// VectorQuantizedVAE_18313740550490
// MI455X (gfx1250) — matrix operand order repaired and checked by reading, not compiled, not run on hardware
//
#include <hip/hip_runtime.h>
#include <hip/hip_bf16.h>
#include <math.h>

typedef __attribute__((ext_vector_type(16))) _Float16 v16h;
typedef __attribute__((ext_vector_type(8)))  _Float16 v8h;
typedef __attribute__((ext_vector_type(8)))  float    v8f;

#define WG 256

// ---------------------------------------------------------------------------
// small prep kernels
// ---------------------------------------------------------------------------
__global__ void k_zero1(float* p) {
  if (threadIdx.x == 0 && blockIdx.x == 0) *p = 0.f;
}

__global__ void k_cast_f16(const float* __restrict__ s, _Float16* __restrict__ d, int n) {
  int i = blockIdx.x * blockDim.x + threadIdx.x;
  if (i < n) d[i] = (_Float16)s[i];
}

// Pack conv weights OIHW (or deconv IOHW) into per-lane WMMA B-fragment order:
// dst[((kChunk*NG + nGroup)*32 + lane)*16 + j] = B[k, n]
//   with k = kChunk*32 + (lane>>4)*16 + j,  n = nGroup*16 + (lane&15)
// so each lane's 16-f16 fragment is one contiguous 32-byte global load.
__global__ void k_pack_w(const float* __restrict__ src, _Float16* __restrict__ dst,
                         int Cin, int Cout, int KH, int KW, int Ktot, int Kpad,
                         int transposed) {
  int t = blockIdx.x * blockDim.x + threadIdx.x;
  if (t >= Kpad * Cout) return;
  int co = t % Cout, kk = t / Cout;
  float v = 0.f;
  if (kk < Ktot) {
    int ci = kk % Cin, rem = kk / Cin;
    int kw = rem % KW, kh = rem / KW;
    v = transposed ? src[((size_t)(ci * Cout + co) * KH + kh) * KW + kw]
                   : src[((size_t)(co * Cin + ci) * KH + kh) * KW + kw];
  }
  int kcIdx = kk >> 5, kr = kk & 31;
  int half = kr >> 4, j = kr & 15;
  int NG = Cout >> 4;
  int nGroup = co >> 4, l16c = co & 15;
  int lane = (half << 4) | l16c;
  dst[(size_t)(((kcIdx * NG + nGroup) << 5) + lane) * 16 + j] = (_Float16)v;
}

// codebook [1024,256] f32 -> fragment-ordered f16 (K=256 -> 8 chunks, N=1024 ->
// 64 groups) + row norms
__global__ void k_cb_prep(const float* __restrict__ cb, _Float16* __restrict__ cbf,
                          float* __restrict__ cbn) {
  int n = blockIdx.x * blockDim.x + threadIdx.x;
  if (n >= 1024) return;
  float s = 0.f;
  int nGroup = n >> 4, l16c = n & 15;
  for (int k = 0; k < 256; ++k) {
    float v = cb[(size_t)n * 256 + k];
    s += v * v;
    int ch = k >> 5, kr = k & 31;
    int half = kr >> 4, j = kr & 15;
    int lane = (half << 4) | l16c;
    cbf[(size_t)(((ch * 64 + nGroup) << 5) + lane) * 16 + j] = (_Float16)v;
  }
  cbn[n] = s;
}

// ---------------------------------------------------------------------------
// implicit-GEMM conv / transposed-conv, geometry fully compile-time.
// WG = 8 waves; block computes 64(M) x 64(N); each wave computes TWO 16x16
// tiles (nT, nT+16) off one shared A fragment -> 2 independent back-to-back
// v_wmma_f32_16x16x32_f16 per 32-K chunk. A tile gathered to LDS (implicit
// im2col); B fragments read straight from fragment-packed global weights.
// ---------------------------------------------------------------------------
template <int IH, int IW, int Cin, int OH, int OW, int Cout,
          int KH, int KW, int STRIDE, int PAD, int KTOT, int KPAD,
          bool TRANSPOSED, bool RELU>
__global__ void __launch_bounds__(WG)
k_conv_gemm(const _Float16* __restrict__ A, const _Float16* __restrict__ Bp,
            const float* __restrict__ bias,
            float* __restrict__ outF32, _Float16* __restrict__ outF16) {
  constexpr int NG = Cout >> 4;
  __shared__ _Float16 As[64 * 32];
  const int t = threadIdx.x;
  const int wave = t >> 5, lane = t & 31;
  const int l16 = lane & 15, half = lane >> 4;
  const int mBase = blockIdx.x * 64;
  const int nBase = blockIdx.y * 64;
  const int mT = (wave >> 1) << 4;   // 0,16,32,48
  const int nT = (wave & 1) << 5;    // 0,32
  const int nGroup0 = (nBase + nT) >> 4;   // two adjacent 16-col groups

  v8f c0 = {0.f, 0.f, 0.f, 0.f, 0.f, 0.f, 0.f, 0.f};
  v8f c1 = {0.f, 0.f, 0.f, 0.f, 0.f, 0.f, 0.f, 0.f};

  for (int kc = 0; kc < KPAD; kc += 32) {
    // ---- stage A tile (64 rows x 32 k): implicit im2col gather ----
#pragma unroll
    for (int i = 0; i < 8; ++i) {
      int e = t + i * WG;
      int r = e >> 5, kk = e & 31;
      int K = kc + kk;
      _Float16 hv = (_Float16)0.f;
      if (KTOT == KPAD || K < KTOT) {
        int ci = K % Cin;
        int rem = K / Cin;
        int kw = rem % KW, kh = rem / KW;
        int m = mBase + r;
        int ox = m % OW; int tmp = m / OW;
        int oy = tmp % OH; int n = tmp / OH;
        int iy, ix; bool ok;
        if (!TRANSPOSED) {
          iy = oy * STRIDE - PAD + kh;
          ix = ox * STRIDE - PAD + kw;
          ok = (iy >= 0) && (iy < IH) && (ix >= 0) && (ix < IW);
        } else {
          int ty = oy + PAD - kh, tx = ox + PAD - kw;
          ok = (ty >= 0) && (tx >= 0) && ((ty % STRIDE) == 0) && ((tx % STRIDE) == 0);
          iy = ty / STRIDE; ix = tx / STRIDE;
          ok = ok && (iy < IH) && (ix < IW);
        }
        if (ok) hv = A[(size_t)((n * IH + iy) * IW + ix) * Cin + ci];
      }
      As[(r << 5) + kk] = hv;
    }
    __syncthreads();
    // ---- A fragment: two contiguous 16B LDS runs per lane (shared by both
    //      WMMAs) ----
    const int row = mT + l16;
    v8h alo = *(const v8h*)&As[(row << 5) + (half << 4)];
    v8h ahi = *(const v8h*)&As[(row << 5) + (half << 4) + 8];
    v16h af = __builtin_shufflevector(alo, ahi,
                                      0, 1, 2, 3, 4, 5, 6, 7,
                                      8, 9, 10, 11, 12, 13, 14, 15);
    // ---- B fragments: contiguous 32B global loads per lane ----
    const size_t bBase = (size_t)((((kc >> 5) * NG + nGroup0) << 5) + lane) * 16;
    v16h bf0 = *(const v16h*)&Bp[bBase];
    v16h bf1 = *(const v16h*)&Bp[bBase + 32 * 16];
    c0 = __builtin_amdgcn_wmma_f32_16x16x32_f16(false, af, false, bf0,
                                                (short)0, c0, false, false);
    c1 = __builtin_amdgcn_wmma_f32_16x16x32_f16(false, af, false, bf1,
                                                (short)0, c1, false, false);
    __syncthreads();
  }
  // ---- epilogue: bias (+ReLU), NHWC stores for both column tiles ----
  const int co0 = nBase + nT + l16;
  const int co1 = co0 + 16;
  const float bv0 = bias[co0];
  const float bv1 = bias[co1];
#pragma unroll
  for (int r = 0; r < 8; ++r) {
    int row = mT + r + (half << 3);
    size_t ob = (size_t)(mBase + row) * Cout;
    float v0 = c0[r] + bv0;
    float v1 = c1[r] + bv1;
    if (RELU) { v0 = fmaxf(v0, 0.f); v1 = fmaxf(v1, 0.f); }
    if (outF32) { outF32[ob + co0] = v0; outF32[ob + co1] = v1; }
    if (outF16) { outF16[ob + co0] = (_Float16)v0; outF16[ob + co1] = (_Float16)v1; }
  }
}

// ---------------------------------------------------------------------------
// fused VQ distance GEMM + argmin. One wave handles 16 rows of z against all
// 1024 codes: dist = ||cb||^2 - 2 z.cb  (||z||^2 dropped: constant per row).
// ---------------------------------------------------------------------------
__global__ void __launch_bounds__(32)
k_vq_argmin(const _Float16* __restrict__ z16, const _Float16* __restrict__ cbf,
            const float* __restrict__ cbn, int* __restrict__ idxOut) {
  __shared__ _Float16 As[16 * 256];
  const int lane = threadIdx.x;
  const int l16 = lane & 15, half = lane >> 4;
  const int mBase = blockIdx.x << 4;
  // vectorized stage: 8KB as 512 x 16B
  {
    const v8h* src = (const v8h*)(z16 + (size_t)mBase * 256);
    v8h* dst = (v8h*)As;
    for (int i = lane; i < 512; i += 32) dst[i] = src[i];
  }
  __syncthreads();

  v16h a[8];
  const _Float16* arow = &As[l16 << 8];
#pragma unroll
  for (int ch = 0; ch < 8; ++ch) {
    v8h lo = *(const v8h*)&arow[(ch << 5) + (half << 4)];
    v8h hi = *(const v8h*)&arow[(ch << 5) + (half << 4) + 8];
    a[ch] = __builtin_shufflevector(lo, hi,
                                    0, 1, 2, 3, 4, 5, 6, 7,
                                    8, 9, 10, 11, 12, 13, 14, 15);
  }

  float best[8]; int bidx[8];
#pragma unroll
  for (int r = 0; r < 8; ++r) { best[r] = 3.4e38f; bidx[r] = 0; }

  for (int nt = 0; nt < 64; ++nt) {
    const int n = (nt << 4) + l16;
    v8f c = {0.f, 0.f, 0.f, 0.f, 0.f, 0.f, 0.f, 0.f};
#pragma unroll
    for (int ch = 0; ch < 8; ++ch) {
      v16h b = *(const v16h*)&cbf[(size_t)((((ch << 6) + nt) << 5) + lane) * 16];
      c = __builtin_amdgcn_wmma_f32_16x16x32_f16(false, a[ch], false, b,
                                                 (short)0, c, false, false);
    }
    const float cn = cbn[n];
#pragma unroll
    for (int r = 0; r < 8; ++r) {
      float d = cn - 2.f * c[r];
      if (d < best[r]) { best[r] = d; bidx[r] = n; }
    }
  }
  // cross-lane reduce within each 16-lane half (rows r / r+8)
#pragma unroll
  for (int r = 0; r < 8; ++r) {
    float bv = best[r]; int bi = bidx[r];
    for (int off = 8; off > 0; off >>= 1) {
      float ov = __shfl_xor(bv, off, 16);
      int   oi = __shfl_xor(bi, off, 16);
      if (ov < bv || (ov == bv && oi < bi)) { bv = ov; bi = oi; }
    }
    if (l16 == 0) idxOut[mBase + r + (half << 3)] = bi;
  }
}

// ---------------------------------------------------------------------------
// e = codebook[idx] (f16 for decoder) + diff = mean((z - e)^2)
// ---------------------------------------------------------------------------
__global__ void __launch_bounds__(256)
k_gather_diff(const int* __restrict__ idx, const float* __restrict__ cb,
              const float* __restrict__ z32, _Float16* __restrict__ e16,
              float* __restrict__ diffOut) {
  __shared__ float red[256];
  const int m = blockIdx.x, c = threadIdx.x;
  const int id = idx[m];
  const float cv = cb[(size_t)id * 256 + c];
  const size_t o = (size_t)m * 256 + c;
  e16[o] = (_Float16)cv;
  float d = z32[o] - cv;
  red[c] = d * d;
  __syncthreads();
  for (int s = 128; s > 0; s >>= 1) {
    if (c < s) red[c] += red[c + s];
    __syncthreads();
  }
  if (c == 0) atomicAdd(diffOut, red[0] * (1.f / (65536.f * 256.f)));
}

// ---------------------------------------------------------------------------
// final 3x3 conv (64->1) + tanh, direct (N=1 too thin for WMMA);
// channel loop vectorized over contiguous 128B NHWC runs.
// ---------------------------------------------------------------------------
__global__ void __launch_bounds__(256)
k_conv3x3_tanh(const _Float16* __restrict__ in, const float* __restrict__ w,
               const float* __restrict__ b, float* __restrict__ out) {
  __shared__ float ws[9 * 64];   // [tap][c]
  const int t = threadIdx.x;
  for (int i = t; i < 576; i += 256) {
    int c2 = i % 64, tap = i / 64;          // source w is [c][3][3]
    ws[tap * 64 + c2] = w[c2 * 9 + tap];
  }
  __syncthreads();
  const int g = blockIdx.x * 256 + t;
  const int x = g & 127, y = (g >> 7) & 127, n = g >> 14;
  float acc = b[0];
  for (int ky = 0; ky < 3; ++ky) {
    int iy = y - 1 + ky; if (iy < 0 || iy > 127) continue;
    for (int kx = 0; kx < 3; ++kx) {
      int ix = x - 1 + kx; if (ix < 0 || ix > 127) continue;
      const v8h* pv = (const v8h*)(in + ((size_t)(n * 128 + iy) * 128 + ix) * 64);
      const float* wt = &ws[(ky * 3 + kx) * 64];
#pragma unroll
      for (int v8i = 0; v8i < 8; ++v8i) {
        v8h hv = pv[v8i];
#pragma unroll
        for (int q = 0; q < 8; ++q)
          acc += (float)hv[q] * wt[v8i * 8 + q];
      }
    }
  }
  out[g] = tanhf(acc);
}

// ---------------------------------------------------------------------------
extern "C" void kernel_launch(void* const* d_in, const int* in_sizes, int n_in,
                              void* d_out, int out_size, void* d_ws, size_t ws_size,
                              hipStream_t stream) {
  (void)in_sizes; (void)n_in; (void)out_size; (void)ws_size;
  const float* x   = (const float*)d_in[0];
  const float* ew1 = (const float*)d_in[1];  const float* eb1 = (const float*)d_in[2];
  const float* ew2 = (const float*)d_in[3];  const float* eb2 = (const float*)d_in[4];
  const float* ew3 = (const float*)d_in[5];  const float* eb3 = (const float*)d_in[6];
  const float* cb  = (const float*)d_in[7];
  const float* dw1 = (const float*)d_in[8];  const float* db1 = (const float*)d_in[9];
  const float* dw2 = (const float*)d_in[10]; const float* db2 = (const float*)d_in[11];
  const float* dw3 = (const float*)d_in[12]; const float* db3 = (const float*)d_in[13];
  float* out = (float*)d_out;

  char* ws = (char*)d_ws; size_t off = 0;
  auto alloc = [&](size_t bytes) -> void* {
    void* p = ws + off; off += (bytes + 255) & ~(size_t)255; return p;
  };
  _Float16* xh  = (_Float16*)alloc((size_t)64 * 128 * 128 * 2);   // x NHWC f16 (C=1)
  _Float16* a1  = (_Float16*)alloc((size_t)262144 * 64 * 2);      // enc1 out
  _Float16* a2  = (_Float16*)alloc((size_t)262144 * 64 * 2);      // enc2 out
  float*    z32 = (float*)   alloc((size_t)65536 * 256 * 4);      // z f32
  _Float16* z16 = (_Float16*)alloc((size_t)65536 * 256 * 2);      // z f16
  _Float16* e16 = (_Float16*)alloc((size_t)65536 * 256 * 2);      // quantized e
  _Float16* d1  = (_Float16*)alloc((size_t)262144 * 64 * 2);      // dec1 out
  _Float16* d2  = (_Float16*)alloc((size_t)1048576 * 64 * 2);     // dec2 out
  _Float16* w1p = (_Float16*)alloc((size_t)32 * 64 * 2);
  _Float16* w2p = (_Float16*)alloc((size_t)1600 * 64 * 2);
  _Float16* w3p = (_Float16*)alloc((size_t)1600 * 256 * 2);
  _Float16* t1p = (_Float16*)alloc((size_t)4096 * 64 * 2);
  _Float16* t2p = (_Float16*)alloc((size_t)1024 * 64 * 2);
  _Float16* cbf = (_Float16*)alloc((size_t)256 * 1024 * 2);
  float*    cbn = (float*)   alloc(1024 * 4);
  int*      idx = (int*)     alloc((size_t)65536 * 4);

  // ---- prep ----
  k_zero1<<<1, 1, 0, stream>>>(out + 1048576);
  k_cast_f16<<<(1048576 + 255) / 256, 256, 0, stream>>>(x, xh, 1048576);
  k_pack_w<<<(32 * 64 + 255) / 256, 256, 0, stream>>>(ew1, w1p, 1, 64, 5, 5, 25, 32, 0);
  k_pack_w<<<(1600 * 64 + 255) / 256, 256, 0, stream>>>(ew2, w2p, 64, 64, 5, 5, 1600, 1600, 0);
  k_pack_w<<<(1600 * 256 + 255) / 256, 256, 0, stream>>>(ew3, w3p, 64, 256, 5, 5, 1600, 1600, 0);
  k_pack_w<<<(4096 * 64 + 255) / 256, 256, 0, stream>>>(dw1, t1p, 256, 64, 4, 4, 4096, 4096, 1);
  k_pack_w<<<(1024 * 64 + 255) / 256, 256, 0, stream>>>(dw2, t2p, 64, 64, 4, 4, 1024, 1024, 1);
  k_cb_prep<<<4, 256, 0, stream>>>(cb, cbf, cbn);

  // ---- encoder ----
  k_conv_gemm<128, 128, 1, 64, 64, 64, 5, 5, 2, 2, 25, 32, false, true>
      <<<dim3(4096, 1), WG, 0, stream>>>(xh, w1p, eb1, nullptr, a1);
  k_conv_gemm<64, 64, 64, 64, 64, 64, 5, 5, 1, 2, 1600, 1600, false, true>
      <<<dim3(4096, 1), WG, 0, stream>>>(a1, w2p, eb2, nullptr, a2);
  k_conv_gemm<64, 64, 64, 32, 32, 256, 5, 5, 2, 2, 1600, 1600, false, false>
      <<<dim3(1024, 4), WG, 0, stream>>>(a2, w3p, eb3, z32, z16);

  // ---- quantize ----
  k_vq_argmin<<<4096, 32, 0, stream>>>(z16, cbf, cbn, idx);
  k_gather_diff<<<65536, 256, 0, stream>>>(idx, cb, z32, e16, out + 1048576);

  // ---- decoder ----
  k_conv_gemm<32, 32, 256, 64, 64, 64, 4, 4, 2, 1, 4096, 4096, true, true>
      <<<dim3(4096, 1), WG, 0, stream>>>(e16, t1p, db1, nullptr, d1);
  k_conv_gemm<64, 64, 64, 128, 128, 64, 4, 4, 2, 1, 1024, 1024, true, true>
      <<<dim3(16384, 1), WG, 0, stream>>>(d1, t2p, db2, nullptr, d2);
  k_conv3x3_tanh<<<4096, 256, 0, stream>>>(d2, dw3, db3, out);
}
